// TumorGATClassifier_22230750724498
// MI455X (gfx1250) — matrix operand order repaired and checked by reading, not compiled, not run on hardware
//
#include <hip/hip_runtime.h>
#include <hip/hip_bf16.h>

// ---------------- model constants (match reference) ----------------
#define HEADS 4
#define HID   64
#define HC    256     // HEADS * HID
#define SLOPE 0.2f

typedef __bf16 bf16_t;
typedef __attribute__((ext_vector_type(16))) __bf16 v16bf;
typedef __attribute__((ext_vector_type(8)))  __bf16 v8bf;
typedef __attribute__((ext_vector_type(8)))  float  v8f;

// ---------------- helpers ----------------
__global__ void cvt_f32_bf16(const float* __restrict__ in, bf16_t* __restrict__ out, long n) {
  long i = (long)blockIdx.x * blockDim.x + threadIdx.x;
  long stride = (long)gridDim.x * blockDim.x;
  for (; i < n; i += stride) out[i] = (bf16_t)in[i];
}

__global__ void fill_value(float* __restrict__ p, float v, long n) {
  long i = (long)blockIdx.x * blockDim.x + threadIdx.x;
  long stride = (long)gridDim.x * blockDim.x;
  for (; i < n; i += stride) p[i] = v;
}

// float atomic max via sign-split int/uint atomics (init must be -inf)
__device__ inline void atomicMaxF(float* addr, float val) {
  if (val >= 0.f) atomicMax((int*)addr, __float_as_int(val));
  else            atomicMin((unsigned int*)addr, __float_as_uint(val));
}

// ---------------- WMMA GEMM:  C[M x Nout] = A[M x K] * B[Nout x K]^T ----------------
// A, B bf16 row-major; C f32 row-major. One wave32 per 16x16 output tile.
// Fragment layouts follow CDNA5 ISA 7.12.2 (16-bit A 16x32; B 32x16; 32-bit C/D 16x16).
__global__ void gemm_wmma_bf16(const bf16_t* __restrict__ A, const bf16_t* __restrict__ B,
                               float* __restrict__ C, int M, int K, int Nout) {
  int wave   = (int)((blockIdx.x * blockDim.x + threadIdx.x) >> 5);
  int lane   = threadIdx.x & 31;
  int tilesN = Nout >> 4;
  int tilesM = (M + 15) >> 4;
  int tm = wave / tilesN;
  int tn = wave - tm * tilesN;
  if (tm >= tilesM) return;

  int half = lane >> 4;      // which K-half this lane carries
  int lr   = lane & 15;      // M index (A) / N index (B, C)

  int arowIdx = tm * 16 + lr;
  if (arowIdx >= M) arowIdx = M - 1;                 // clamp (store is masked)
  const bf16_t* arow = A + (size_t)arowIdx * K + half * 8;
  const bf16_t* brow = B + (size_t)(tn * 16 + lr) * K + half * 8;

  v8f acc = {};
  for (int k0 = 0; k0 < K; k0 += 32) {
    __builtin_prefetch(arow + k0 + 128, 0, 1);       // global_prefetch_b8
    // A 16x32 bf16: lanes<16 carry K {0..7, 16..23}; lanes>=16 carry {8..15, 24..31}
    v8bf alo = *(const v8bf*)(arow + k0);
    v8bf ahi = *(const v8bf*)(arow + k0 + 16);
    v16bf a;
#pragma unroll
    for (int i = 0; i < 8; ++i) { a[i] = alo[i]; a[i + 8] = ahi[i]; }
    // B 32x16 bf16: lanes<16 carry K {0..7, 16..23} of column lr; lanes>=16 carry {8..15, 24..31}
    v8bf blo = *(const v8bf*)(brow + k0);
    v8bf bhi = *(const v8bf*)(brow + k0 + 16);
    v16bf b;
#pragma unroll
    for (int i = 0; i < 8; ++i) { b[i] = blo[i]; b[i + 8] = bhi[i]; }
    acc = __builtin_amdgcn_wmma_f32_16x16x32_bf16(
        /*neg_a=*/false, a, /*neg_b=*/false, b,
        /*c_mod=*/(short)0, acc, /*reuse_a=*/false, /*reuse_b=*/false);
  }
  // C/D layout: VGPR r -> row (half*8 + r), column lr
  int col   = tn * 16 + lr;
  int rbase = tm * 16 + half * 8;
#pragma unroll
  for (int r = 0; r < 8; ++r) {
    int row = rbase + r;
    if (row < M) C[(size_t)row * Nout + col] = acc[r];
  }
}

// ---------------- attention coefficients: al = sum_c h[n,head,c] * a[head,c] ----------------
__global__ void att_coeff(const float* __restrict__ h, const float* __restrict__ a_s,
                          const float* __restrict__ a_d, float* __restrict__ als,
                          float* __restrict__ ald, int N_) {
  int idx = blockIdx.x * blockDim.x + threadIdx.x;
  if (idx >= N_ * HEADS) return;
  int node = idx / HEADS, head = idx % HEADS;
  const float* hp  = h  + (size_t)node * HC + head * HID;
  const float* asp = a_s + head * HID;
  const float* adp = a_d + head * HID;
  float ss = 0.f, sd = 0.f;
#pragma unroll 4
  for (int c = 0; c < HID; ++c) { float v = hp[c]; ss += v * asp[c]; sd += v * adp[c]; }
  als[idx] = ss;
  ald[idx] = sd;
}

// ---------------- edge pipeline ----------------
__global__ void build_edges(const int* __restrict__ ei, int* __restrict__ src,
                            int* __restrict__ dst, int E_, int N_) {
  int i = blockIdx.x * blockDim.x + threadIdx.x;
  int tot = E_ + N_;
  if (i >= tot) return;
  if (i < E_) { src[i] = ei[i]; dst[i] = ei[E_ + i]; }     // edge_index rows 0 / 1
  else        { int v = i - E_; src[i] = v; dst[i] = v; }  // self loops
}

__global__ void edge_logits_max(const int* __restrict__ src, const int* __restrict__ dst,
                                const float* __restrict__ als, const float* __restrict__ ald,
                                float* __restrict__ ebuf, float* __restrict__ mbuf, int Etot) {
  int idx = blockIdx.x * blockDim.x + threadIdx.x;
  if (idx >= Etot * HEADS) return;
  int edge = idx >> 2, head = idx & (HEADS - 1);
  int s = src[edge], d = dst[edge];
  float e = als[s * HEADS + head] + ald[d * HEADS + head];
  e = (e > 0.f) ? e : SLOPE * e;                  // leaky relu
  ebuf[idx] = e;
  atomicMaxF(&mbuf[d * HEADS + head], e);
}

__global__ void edge_exp_sum(const int* __restrict__ dst, float* __restrict__ ebuf,
                             const float* __restrict__ mbuf, float* __restrict__ dbuf, int Etot) {
  int idx = blockIdx.x * blockDim.x + threadIdx.x;
  if (idx >= Etot * HEADS) return;
  int edge = idx >> 2, head = idx & (HEADS - 1);
  int d = dst[edge];
  float ex = __expf(ebuf[idx] - mbuf[d * HEADS + head]);
  ebuf[idx] = ex;
  atomicAdd(&dbuf[d * HEADS + head], ex);
}

// one block (256 threads) per edge: coalesced gather of h[src,:] and atomic scatter to out[dst,:]
__global__ void edge_aggregate(const int* __restrict__ src, const int* __restrict__ dst,
                               const float* __restrict__ ebuf, const float* __restrict__ dbuf,
                               const float* __restrict__ h, float* __restrict__ out, int Etot) {
  int edge = blockIdx.x;
  if (edge >= Etot) return;
  __shared__ float s_alpha[HEADS];
  __shared__ int   s_s, s_d;
  if (threadIdx.x == 0) { s_s = src[edge]; s_d = dst[edge]; }
  __syncthreads();
  if (threadIdx.x < HEADS)
    s_alpha[threadIdx.x] =
        ebuf[edge * HEADS + threadIdx.x] / (dbuf[s_d * HEADS + threadIdx.x] + 1e-16f);
  __syncthreads();
  int ch = threadIdx.x;                 // 0..255
  int head = ch >> 6;                   // ch / HID
  float v = h[(size_t)s_s * HC + ch] * s_alpha[head];
  atomicAdd(&out[(size_t)s_d * HC + ch], v);
}

__global__ void add_bias_relu(float* __restrict__ out, const float* __restrict__ bias,
                              int N_, int relu) {
  long idx = (long)blockIdx.x * blockDim.x + threadIdx.x;
  if (idx >= (long)N_ * HC) return;
  int ch = (int)(idx & (HC - 1));
  float v = out[idx] + bias[ch];
  if (relu) v = fmaxf(v, 0.f);
  out[idx] = v;
}

// ---------------- pooling + MLP head ----------------
__global__ void pool_accum(const float* __restrict__ act, const int* __restrict__ batch,
                           float* __restrict__ pooled, float* __restrict__ cnt, int N_) {
  long idx = (long)blockIdx.x * blockDim.x + threadIdx.x;
  if (idx >= (long)N_ * HC) return;
  int node = (int)(idx >> 8);           // HC == 256
  int ch   = (int)(idx & (HC - 1));
  int g = batch[node];
  atomicAdd(&pooled[(size_t)g * HC + ch], act[idx]);
  if (ch == 0) atomicAdd(&cnt[g], 1.f);
}

__global__ void mlp_fc1(const float* __restrict__ pooled, const float* __restrict__ cnt,
                        const float* __restrict__ w, const float* __restrict__ b,
                        float* __restrict__ z) {
  int g = blockIdx.x;
  int o = threadIdx.x;                  // HID threads
  float scale = 1.f / fmaxf(cnt[g], 1.f);
  float acc = b[o];
  const float* pg = pooled + (size_t)g * HC;
  const float* wo = w + (size_t)o * HC;
#pragma unroll 4
  for (int k = 0; k < HC; ++k) acc += pg[k] * scale * wo[k];
  z[g * HID + o] = fmaxf(acc, 0.f);
}

__global__ void mlp_fc2(const float* __restrict__ z, const float* __restrict__ w,
                        const float* __restrict__ b, float* __restrict__ out, int G_, int NC_) {
  int idx = blockIdx.x * blockDim.x + threadIdx.x;
  if (idx >= G_ * NC_) return;
  int g = idx / NC_, c = idx % NC_;
  float acc = b[c];
  const float* zg = z + (size_t)g * HID;
  const float* wc = w + (size_t)c * HID;
#pragma unroll 4
  for (int k = 0; k < HID; ++k) acc += zg[k] * wc[k];
  out[idx] = acc;
}

// ---------------- launcher ----------------
extern "C" void kernel_launch(void* const* d_in, const int* in_sizes, int n_in,
                              void* d_out, int out_size, void* d_ws, size_t ws_size,
                              hipStream_t stream) {
  (void)n_in; (void)ws_size;
  const float* x     = (const float*)d_in[0];
  const int*   ei    = (const int*)d_in[1];
  const int*   batch = (const int*)d_in[2];
  const float* W[3]  = {(const float*)d_in[3], (const float*)d_in[7],  (const float*)d_in[11]};
  const float* As[3] = {(const float*)d_in[4], (const float*)d_in[8],  (const float*)d_in[12]};
  const float* Ad[3] = {(const float*)d_in[5], (const float*)d_in[9],  (const float*)d_in[13]};
  const float* Bs[3] = {(const float*)d_in[6], (const float*)d_in[10], (const float*)d_in[14]};
  const float* fc1w = (const float*)d_in[15];
  const float* fc1b = (const float*)d_in[16];
  const float* fc2w = (const float*)d_in[17];
  const float* fc2b = (const float*)d_in[18];

  const int N_   = in_sizes[2];
  const int E_   = in_sizes[1] / 2;
  const int IN_  = in_sizes[0] / N_;
  const int Etot = E_ + N_;
  const int NC_  = in_sizes[18];
  const int G_   = out_size / NC_;

  // workspace carving (256B aligned slabs)
  char* p = (char*)d_ws;
  auto carve = [&](size_t bytes) { char* r = p; p += (bytes + 255) & ~(size_t)255; return r; };
  bf16_t* actbf = (bf16_t*)carve((size_t)N_ * HC * sizeof(bf16_t));
  bf16_t* wbf   = (bf16_t*)carve((size_t)HC * HC * sizeof(bf16_t));
  float*  hbuf  = (float*)carve((size_t)N_ * HC * sizeof(float));
  float*  aggA  = (float*)carve((size_t)N_ * HC * sizeof(float));
  float*  aggB  = (float*)carve((size_t)N_ * HC * sizeof(float));
  float*  als   = (float*)carve((size_t)N_ * HEADS * sizeof(float));
  float*  ald   = (float*)carve((size_t)N_ * HEADS * sizeof(float));
  float*  mbuf  = (float*)carve((size_t)N_ * HEADS * sizeof(float));
  float*  dbuf  = (float*)carve((size_t)N_ * HEADS * sizeof(float));
  float*  ebuf  = (float*)carve((size_t)Etot * HEADS * sizeof(float));
  int*    src32 = (int*)carve((size_t)Etot * sizeof(int));
  int*    dst32 = (int*)carve((size_t)Etot * sizeof(int));
  float*  pooled= (float*)carve((size_t)G_ * HC * sizeof(float));
  float*  cnt   = (float*)carve((size_t)G_ * sizeof(float));
  float*  zbuf  = (float*)carve((size_t)G_ * HID * sizeof(float));

  build_edges<<<(Etot + 255) / 256, 256, 0, stream>>>(ei, src32, dst32, E_, N_);

  const float* cur_in = x;
  int K = IN_;
  float* outs[3] = {aggA, aggB, aggA};
  for (int l = 0; l < 3; ++l) {
    // bf16 staging for matrix cores
    cvt_f32_bf16<<<4096, 256, 0, stream>>>(cur_in, actbf, (long)N_ * K);
    cvt_f32_bf16<<<256, 256, 0, stream>>>(W[l], wbf, (long)HC * K);

    // WMMA GEMM: hbuf = act @ W^T  (bias deferred to post-aggregation, per GATConv)
    int tilesM = (N_ + 15) / 16, tilesN = HC / 16;
    int waves = tilesM * tilesN;                         // one wave per tile
    gemm_wmma_bf16<<<(waves + 7) / 8, 256, 0, stream>>>(actbf, wbf, hbuf, N_, K, HC);

    att_coeff<<<(N_ * HEADS + 255) / 256, 256, 0, stream>>>(hbuf, As[l], Ad[l], als, ald, N_);

    hipMemsetAsync(dbuf, 0, (size_t)N_ * HEADS * sizeof(float), stream);
    hipMemsetAsync(outs[l], 0, (size_t)N_ * HC * sizeof(float), stream);
    fill_value<<<(N_ * HEADS + 255) / 256, 256, 0, stream>>>(mbuf, -__builtin_inff(),
                                                             (long)N_ * HEADS);

    edge_logits_max<<<(Etot * HEADS + 255) / 256, 256, 0, stream>>>(src32, dst32, als, ald,
                                                                    ebuf, mbuf, Etot);
    edge_exp_sum<<<(Etot * HEADS + 255) / 256, 256, 0, stream>>>(dst32, ebuf, mbuf, dbuf, Etot);
    edge_aggregate<<<Etot, HC, 0, stream>>>(src32, dst32, ebuf, dbuf, hbuf, outs[l], Etot);
    add_bias_relu<<<(int)(((long)N_ * HC + 255) / 256), 256, 0, stream>>>(outs[l], Bs[l], N_,
                                                                          l < 2 ? 1 : 0);
    cur_in = outs[l];
    K = HC;
  }

  // global mean pool + MLP head
  hipMemsetAsync(pooled, 0, (size_t)G_ * HC * sizeof(float), stream);
  hipMemsetAsync(cnt, 0, (size_t)G_ * sizeof(float), stream);
  pool_accum<<<(int)(((long)N_ * HC + 255) / 256), 256, 0, stream>>>(outs[2], batch, pooled,
                                                                     cnt, N_);
  mlp_fc1<<<G_, HID, 0, stream>>>(pooled, cnt, fc1w, fc1b, zbuf);
  mlp_fc2<<<(G_ * NC_ + 127) / 128, 128, 0, stream>>>(zbuf, fc2w, fc2b, (float*)d_out, G_, NC_);
}
